// ModulateAttention_46497315947272
// MI455X (gfx1250) — matrix operand order repaired and checked by reading, not compiled, not run on hardware
//
#include <hip/hip_runtime.h>

typedef _Float16 f16;
typedef __attribute__((ext_vector_type(16))) _Float16 v16h;
typedef __attribute__((ext_vector_type(8)))  _Float16 v8h;
typedef __attribute__((ext_vector_type(4)))  _Float16 v4h;
typedef __attribute__((ext_vector_type(8)))  float    v8f;
typedef __attribute__((ext_vector_type(4)))  float    v4f;
typedef __attribute__((ext_vector_type(4)))  int      v4i;

#define B_SZ   4
#define T_LEN  2048
#define L_LEN  512
#define CH     1024
#define NH     16
#define HD     64
static constexpr float SCALE = 0.3535533905932738f; // 64^-0.25

#if __has_builtin(__builtin_amdgcn_global_load_async_to_lds_b128)
#define HAS_ASYNC_LDS 1
typedef __attribute__((address_space(1))) v4i gv4i;
typedef __attribute__((address_space(3))) v4i lv4i;
#else
#define HAS_ASYNC_LDS 0
#endif

// lane-xor reduce helper via ds_swizzle (group-of-32, xor_mask=m, and_mask=0x1f)
#define SWZ(v, m) __builtin_bit_cast(float, \
    __builtin_amdgcn_ds_swizzle(__builtin_bit_cast(int, (v)), (((m) << 10) | 0x1f)))

__device__ inline v8f wmma16(v16h a, v16h b, v8f c) {
  return __builtin_amdgcn_wmma_f32_16x16x32_f16(false, a, false, b, (short)0, c,
                                                false, false);
}

// 16 contiguous f16 -> v16h (two b128 loads; works for global and LDS)
__device__ inline v16h load16h(const f16* p) {
  v8h lo = *(const v8h*)p;
  v8h hi = *(const v8h*)(p + 8);
  v16h r;
#pragma unroll
  for (int i = 0; i < 8; i++) { r[i] = lo[i]; r[8 + i] = hi[i]; }
  return r;
}

// A-fragment per-lane data: 8 halfs at p, 8 halfs at p+16 (ISA 16-bit A layout)
__device__ inline v16h load_a16(const f16* p) {
  v8h lo = *(const v8h*)p;
  v8h hi = *(const v8h*)(p + 16);
  v16h r;
#pragma unroll
  for (int i = 0; i < 8; i++) { r[i] = lo[i]; r[8 + i] = hi[i]; }
  return r;
}

// 16B global -> LDS copy: async (CDNA5 GLOBAL_LOAD_ASYNC_TO_LDS_B128) or sync
__device__ inline void cp16B(const f16* g, f16* l) {
#if HAS_ASYNC_LDS
  __builtin_amdgcn_global_load_async_to_lds_b128((gv4i*)g, (lv4i*)l, 0, 0);
#else
  *(v8h*)l = *(const v8h*)g;
#endif
}

__device__ inline void wait_stage_prefetched() {  // <=1 prefetched stage (2 ops)
#if HAS_ASYNC_LDS
#if __has_builtin(__builtin_amdgcn_s_wait_asynccnt)
  __builtin_amdgcn_s_wait_asynccnt(2);
#else
  asm volatile("s_wait_asynccnt 0x2" ::: "memory");
#endif
#else
  asm volatile("s_wait_dscnt 0x0" ::: "memory");
#endif
}
__device__ inline void wait_stage_all() {
#if HAS_ASYNC_LDS
#if __has_builtin(__builtin_amdgcn_s_wait_asynccnt)
  __builtin_amdgcn_s_wait_asynccnt(0);
#else
  asm volatile("s_wait_asynccnt 0x0" ::: "memory");
#endif
#else
  asm volatile("s_wait_dscnt 0x0" ::: "memory");
#endif
}
__device__ inline void wait_lds() {
  asm volatile("s_wait_dscnt 0x0" ::: "memory");
}

// ---------------------------------------------------------------------------
// transpose inpt (B,C,T) f32 -> xh (B*T, C) f16
// ---------------------------------------------------------------------------
__global__ __launch_bounds__(256) void transpose_inpt_kernel(
    const float* __restrict__ inpt, f16* __restrict__ xh) {
  __shared__ float tile[32][33];
  int b = blockIdx.z;
  int c0 = blockIdx.y * 32;
  int t0 = blockIdx.x * 32;
  int tx = threadIdx.x & 31;
  int ty = threadIdx.x >> 5;
  const float* src = inpt + ((size_t)b * CH + c0) * T_LEN + t0;
#pragma unroll
  for (int k = 0; k < 4; k++)
    tile[ty + 8 * k][tx] = src[(size_t)(ty + 8 * k) * T_LEN + tx];
  __syncthreads();
  f16* dst = xh + ((size_t)b * T_LEN + t0) * CH + c0;
#pragma unroll
  for (int k = 0; k < 4; k++)
    dst[(size_t)(ty + 8 * k) * CH + tx] = (f16)tile[tx][ty + 8 * k];
}

// ---------------------------------------------------------------------------
// f32 -> f16 bulk convert (vector of 4 per thread)
// ---------------------------------------------------------------------------
__global__ __launch_bounds__(256) void cvt4_kernel(
    const float* __restrict__ src, f16* __restrict__ dst, int n4) {
  int i = blockIdx.x * 256 + threadIdx.x;
  if (i < n4) {
    v4f v = *(const v4f*)(src + (size_t)i * 4);
    v4h h;
#pragma unroll
    for (int k = 0; k < 4; k++) h[k] = (f16)v[k];
    *(v4h*)(dst + (size_t)i * 4) = h;
  }
}

// ---------------------------------------------------------------------------
// Projection: Y = X(f16, MxC) @ Wh(f16, NxC)^T + bias; 16x64 strip per wave.
// mode 0: Y0[m*N+n] = f16((acc+b)*SCALE)                       (Q)
// mode 1: n<1024 -> Y0[m*1024+n] = f16((acc+b)*SCALE)  (K row-major)
//         n>=1024 -> Y1[(n-1024)*L + m] = f16(acc+b)   (V transposed)
// ---------------------------------------------------------------------------
__global__ __launch_bounds__(256) void proj_kernel(
    const f16* __restrict__ X, const f16* __restrict__ Wh,
    const float* __restrict__ bias, f16* __restrict__ Y0, f16* __restrict__ Y1,
    int M, int N, int mode) {
  int lane = threadIdx.x & 31;
  int wave = threadIdx.x >> 5;
  int strips_n = N >> 6;
  int tile = blockIdx.x * 8 + wave;       // grids sized exactly
  int mt = tile / strips_n, ns = tile % strips_n;
  int g = lane >> 4, ln = lane & 15;
  const f16* xrow = X + (size_t)(mt * 16 + ln) * CH;
  const f16* w0 = Wh + (size_t)(ns * 64 + ln) * CH;
  v8f acc[4] = {};
#pragma unroll 2
  for (int c0 = 0; c0 < CH; c0 += 32) {
    v16h a = load16h(xrow + c0 + g * 16);
#pragma unroll
    for (int i = 0; i < 4; i++)
      acc[i] = wmma16(a, load16h(w0 + (size_t)i * 16 * CH + c0 + g * 16), acc[i]);
  }
  int n0 = ns * 64;
#pragma unroll
  for (int i = 0; i < 4; i++) {
    int n = n0 + i * 16 + ln;
    float bv = bias[n];
#pragma unroll
    for (int r = 0; r < 8; r++) {
      int m = mt * 16 + r + 8 * g;
      float v = acc[i][r] + bv;
      if (mode == 0)       Y0[(size_t)m * N + n] = (f16)(v * SCALE);
      else if (n0 < CH)    Y0[(size_t)m * CH + n] = (f16)(v * SCALE);
      else                 Y1[(size_t)(n - CH) * L_LEN + m] = (f16)v;
    }
  }
}

// ---------------------------------------------------------------------------
// Flash cross-attention. Block = one head x 8 row-tiles (8 waves share K/V).
// K/V chunks (32 keys) staged to LDS with async b128 copies, double-buffered.
// ---------------------------------------------------------------------------
__global__ __launch_bounds__(256) void attn_kernel(
    const f16* __restrict__ Qh, const f16* __restrict__ Kh,
    const f16* __restrict__ Vt, f16* __restrict__ Oh) {
  __shared__ f16 kbuf[2][32 * 64];   //  8 KB: [key][d]
  __shared__ f16 vbuf[2][64 * 32];   //  8 KB: [d][key]
  __shared__ f16 pbuf[8][16 * 32];   //  8 KB: per-wave P tile
  int tid = threadIdx.x;
  int lane = tid & 31;
  int wave = tid >> 5;
  int h = blockIdx.x & 15;            // head
  int mt = (blockIdx.x >> 4) * 8 + wave;  // row tile over B*T/16 = 512
  int g = lane >> 4, ln = lane & 15;
  f16* pb = pbuf[wave];

  const f16* qrow = Qh + (size_t)(mt * 16 + ln) * CH + h * HD;
  v16h qa0 = load16h(qrow + 0 + g * 16);
  v16h qa1 = load16h(qrow + 32 + g * 16);

  const f16* kg = Kh + h * HD;                  // + (j0+row)*CH
  const f16* vg = Vt + (size_t)h * HD * L_LEN;  // + d*L_LEN + j0

  int krow = tid >> 3, kco = (tid & 7) * 8;     // 32 rows x 64 halfs
  int vrow = tid >> 2, vco = (tid & 3) * 8;     // 64 rows x 32 halfs
  auto stage = [&](int jc, int bi) {
    int j0 = jc * 32;
    cp16B(kg + (size_t)(j0 + krow) * CH + kco, &kbuf[bi][krow * 64 + kco]);
    cp16B(vg + (size_t)vrow * L_LEN + j0 + vco, &vbuf[bi][vrow * 32 + vco]);
  };

  v8f o[4] = {};
  float mrow[8], lrow[8];
#pragma unroll
  for (int r = 0; r < 8; r++) { mrow[r] = -3.0e30f; lrow[r] = 0.0f; }

  stage(0, 0);
  for (int jc = 0; jc < 16; jc++) {
    int cur = jc & 1;
    if (jc + 1 < 16) {
      stage(jc + 1, cur ^ 1);
      wait_stage_prefetched();   // this wave's chunk jc landed
    } else {
      wait_stage_all();
    }
    __syncthreads();             // every wave's portion of chunk jc landed

    const f16* kb = kbuf[cur];
    const f16* vb = vbuf[cur];
    v16h kb00 = load16h(kb + (ln)      * 64 + 0  + g * 16);
    v16h kb01 = load16h(kb + (ln)      * 64 + 32 + g * 16);
    v16h kb10 = load16h(kb + (16 + ln) * 64 + 0  + g * 16);
    v16h kb11 = load16h(kb + (16 + ln) * 64 + 32 + g * 16);

    v8f z = {};
    v8f s0 = wmma16(qa0, kb00, z); s0 = wmma16(qa1, kb01, s0);
    v8f s1 = wmma16(qa0, kb10, z); s1 = wmma16(qa1, kb11, s1);

    float alpha[8];
#pragma unroll
    for (int r = 0; r < 8; r++) {
      float mx = fmaxf(s0[r], s1[r]);
      mx = fmaxf(mx, SWZ(mx, 1));
      mx = fmaxf(mx, SWZ(mx, 2));
      mx = fmaxf(mx, SWZ(mx, 4));
      mx = fmaxf(mx, SWZ(mx, 8));
      float mnew = fmaxf(mrow[r], mx);
      float a = __expf(mrow[r] - mnew);
      float p0 = __expf(s0[r] - mnew);
      float p1 = __expf(s1[r] - mnew);
      s0[r] = p0; s1[r] = p1;
      float rs = p0 + p1;
      rs += SWZ(rs, 1);
      rs += SWZ(rs, 2);
      rs += SWZ(rs, 4);
      rs += SWZ(rs, 8);
      lrow[r] = lrow[r] * a + rs;
      mrow[r] = mnew;
      alpha[r] = a;
    }
#pragma unroll
    for (int r = 0; r < 8; r++) {
      o[0][r] *= alpha[r]; o[1][r] *= alpha[r];
      o[2][r] *= alpha[r]; o[3][r] *= alpha[r];
    }
    // accumulator layout -> A-fragment layout through per-wave LDS tile
#pragma unroll
    for (int r = 0; r < 8; r++) {
      int m = r + 8 * g;
      pb[m * 32 + ln]      = (f16)s0[r];
      pb[m * 32 + 16 + ln] = (f16)s1[r];
    }
    wait_lds();                          // in-order LDS within wave
    v16h pa = load16h(pb + ln * 32 + g * 16);
#pragma unroll
    for (int ct = 0; ct < 4; ct++)
      o[ct] = wmma16(pa, load16h(vb + (ct * 16 + ln) * 32 + g * 16), o[ct]);
    __syncthreads();             // protect double buffer before next overwrite
  }

#pragma unroll
  for (int r = 0; r < 8; r++) {
    float inv = 1.0f / lrow[r];
    size_t row = (size_t)(mt * 16 + r + 8 * g) * CH + h * HD + ln;
    Oh[row]      = (f16)(o[0][r] * inv);
    Oh[row + 16] = (f16)(o[1][r] * inv);
    Oh[row + 32] = (f16)(o[2][r] * inv);
    Oh[row + 48] = (f16)(o[3][r] * inv);
  }
}

// ---------------------------------------------------------------------------
// Out projection + bias + residual -> (B, C, T) f32; 16x64 strip per wave.
// ---------------------------------------------------------------------------
__global__ __launch_bounds__(256) void out_proj_kernel(
    const f16* __restrict__ Oh, const f16* __restrict__ Wh,
    const float* __restrict__ bout, const float* __restrict__ inpt,
    float* __restrict__ outp) {
  int lane = threadIdx.x & 31;
  int wave = threadIdx.x >> 5;
  int tile = blockIdx.x * 8 + wave;   // 8192 strips, exact
  int mt = tile >> 4;                 // 512 row tiles
  int ns = tile & 15;                 // 16 col strips of 64
  int g = lane >> 4, ln = lane & 15;
  const f16* xrow = Oh + (size_t)(mt * 16 + ln) * CH;
  const f16* w0 = Wh + (size_t)(ns * 64 + ln) * CH;
  v8f acc[4] = {};
#pragma unroll 2
  for (int c0 = 0; c0 < CH; c0 += 32) {
    v16h a = load16h(xrow + c0 + g * 16);
#pragma unroll
    for (int i = 0; i < 4; i++)
      acc[i] = wmma16(a, load16h(w0 + (size_t)i * 16 * CH + c0 + g * 16), acc[i]);
  }
  int m0 = mt * 16;
  int b = m0 >> 11;                   // T_LEN = 2048
  int tl = (m0 & 2047) + 8 * g;
#pragma unroll
  for (int i = 0; i < 4; i++) {
    int c = ns * 64 + i * 16 + ln;
    float bv = bout[c];
    size_t base = ((size_t)b * CH + c) * T_LEN + tl;
    const float* ip = inpt + base;
    float* op = outp + base;
    v4f r0, r1;
#pragma unroll
    for (int k = 0; k < 4; k++) {
      r0[k] = acc[i][k]     + bv + ip[k];
      r1[k] = acc[i][4 + k] + bv + ip[4 + k];
    }
    *(v4f*)op       = r0;
    *(v4f*)(op + 4) = r1;
  }
}

// ---------------------------------------------------------------------------
extern "C" void kernel_launch(void* const* d_in, const int* in_sizes, int n_in,
                              void* d_out, int out_size, void* d_ws,
                              size_t ws_size, hipStream_t stream) {
  const float* inpt  = (const float*)d_in[0];
  const float* aux   = (const float*)d_in[1];
  const float* q_w   = (const float*)d_in[2];
  const float* q_b   = (const float*)d_in[3];
  const float* kv_w  = (const float*)d_in[4];
  const float* kv_b  = (const float*)d_in[5];
  const float* out_w = (const float*)d_in[6];
  const float* out_b = (const float*)d_in[7];
  float* outp = (float*)d_out;

  char* ws = (char*)d_ws;
  const size_t BT = (size_t)B_SZ * T_LEN;                   // 8192
  f16* xh    = (f16*)ws; ws += BT * CH * sizeof(f16);       // 16 MB
  f16* auxh  = (f16*)ws; ws += (size_t)L_LEN * CH * sizeof(f16);
  f16* Qh    = (f16*)ws; ws += BT * CH * sizeof(f16);       // 16 MB
  f16* Kh    = (f16*)ws; ws += (size_t)L_LEN * CH * sizeof(f16);
  f16* Vt    = (f16*)ws; ws += (size_t)CH * L_LEN * sizeof(f16);
  f16* Oh    = (f16*)ws; ws += BT * CH * sizeof(f16);       // 16 MB
  f16* qwh   = (f16*)ws; ws += (size_t)CH * CH * sizeof(f16);
  f16* kvwh  = (f16*)ws; ws += (size_t)2 * CH * CH * sizeof(f16);
  f16* owh   = (f16*)ws; ws += (size_t)CH * CH * sizeof(f16);

  transpose_inpt_kernel<<<dim3(T_LEN / 32, CH / 32, B_SZ), 256, 0, stream>>>(inpt, xh);
  cvt4_kernel<<<(L_LEN * CH / 4) / 256, 256, 0, stream>>>(aux, auxh, L_LEN * CH / 4);
  cvt4_kernel<<<(CH * CH / 4) / 256, 256, 0, stream>>>(q_w, qwh, CH * CH / 4);
  cvt4_kernel<<<(2 * CH * CH / 4) / 256, 256, 0, stream>>>(kv_w, kvwh, 2 * CH * CH / 4);
  cvt4_kernel<<<(CH * CH / 4) / 256, 256, 0, stream>>>(out_w, owh, CH * CH / 4);

  // KV: 32 row-tiles x 32 strips = 1024 wave-tasks -> 128 blocks
  proj_kernel<<<128, 256, 0, stream>>>(auxh, kvwh, kv_b, Kh, Vt, L_LEN, 2 * CH, 1);
  // Q: 512 x 16 = 8192 wave-tasks -> 1024 blocks
  proj_kernel<<<1024, 256, 0, stream>>>(xh, qwh, q_b, Qh, nullptr, (int)BT, CH, 0);
  // Attention: 16 heads x 64 tile-groups = 1024 blocks
  attn_kernel<<<1024, 256, 0, stream>>>(Qh, Kh, Vt, Oh);
  // Out projection: 8192 strips -> 1024 blocks
  out_proj_kernel<<<1024, 256, 0, stream>>>(Oh, owh, out_b, inpt, outp);
}
